// MultiQueryAttention_66692252172514
// MI455X (gfx1250) — compile-verified, matrix operand order checked by reading, not run on hardware
//
#include <hip/hip_runtime.h>
#include <hip/hip_bf16.h>

// ---------------------------------------------------------------------------
// MultiQueryAttention for MI455X (gfx1250), wave32 + WMMA.
//
// Math reduction: the reference's scores einsum sums over heads, so
//   qs[t,d]  = x[t,:] . (scale * sum_h Wq[:,d,h]) + scale*sum_h bq[d,h]
//   k[t,e]   = x[t,:] . Wk[:,e] + bk[e]
//   v[t,e]   = x[t,:] . Wv[:,e] + bv[e]
//   p        = softmax_e(qs[d] * k[e])            (rank-1 per token)
//   out[t, d*32+h] = (sum_e p*v[e]) for all h     (head broadcast)
//
// Heavy compute: X(16384x4096) @ Wb(4096x384) with v_wmma_f32_16x16x32_bf16.
// Wb (3MB bf16, fragment-native blocks) stays L2-resident; X streams through
// LDS once (268MB read); output written once with NT stores (268MB write).
// Roofline ~600MB @ 23.3TB/s ≈ 26us; WMMA (51.5 GFLOP) is far from the
// bottleneck as long as loads are pipelined (B double-buffered below).
// ---------------------------------------------------------------------------

#define EMB_DIM 4096
#define QKV_DIM 128
#define N_HEADS 32
#define N_TOT   384            // 128 qs | 128 k | 128 v
#define SCALE   0.08838834764831845f   // 1/sqrt(128)

typedef __attribute__((ext_vector_type(16))) __bf16 v16bf;
typedef __attribute__((ext_vector_type(8)))  float  v8f;
typedef __attribute__((ext_vector_type(4)))  float  v4f;   // clang-native for NT stores

union Frag16 {                 // 8 VGPRs = 16 bf16 (A or B operand of 16x16x32)
    uint4  q[2];
    v16bf  v;
};

__device__ __forceinline__ unsigned short f2bf(float f) {
    unsigned u = __builtin_bit_cast(unsigned, f);
    u += 0x7FFFu + ((u >> 16) & 1u);       // round-to-nearest-even
    return (unsigned short)(u >> 16);
}

// Fragment index mapping for 16-bit operands (ISA 7.12.2):
// lane L: slot = L%16 (M for A, N for B); half = L/16.
// bf16 element p (0..15) within a lane's 32 bytes maps to
//   K(p, half) = 16*(p/8) + 8*half + (p%8)
// so each lane's data is two contiguous 16-byte runs in K -> two b128 loads.

// ---------------------------------------------------------------------------
// Kernel 1: build Wb bf16 in fragment-native 32x16 blocks:
//   Wb[((nt*128)+ks)*512 + lane*16 + p]  <-  B[k = ks*32 + K(p,half)][n = nt*16 + L%16]
// Columns 0..127 = scale * sum_h Wq ; 128..255 = Wk ; 256..383 = Wv.
// ---------------------------------------------------------------------------
__global__ __launch_bounds__(256) void prep_wb(const float* __restrict__ Wq,
                                               const float* __restrict__ Wk,
                                               const float* __restrict__ Wv,
                                               unsigned short* __restrict__ Wb) {
    const int tid     = blockIdx.x * 256 + threadIdx.x;   // 0 .. 4096*384-1
    const int blockId = tid >> 9;                          // (nt,ks) block
    const int within  = tid & 511;
    const int nt   = blockId / 128;
    const int ks   = blockId % 128;
    const int lane = within >> 4;
    const int p    = within & 15;
    const int half = lane >> 4;
    const int kk   = 16 * (p >> 3) + 8 * half + (p & 7);
    const int k    = ks * 32 + kk;
    const int n    = nt * 16 + (lane & 15);

    float val;
    if (n < 128) {
        const float* src = Wq + ((size_t)k * QKV_DIM + n) * N_HEADS;
        float s = 0.f;
        #pragma unroll
        for (int h = 0; h < N_HEADS; ++h) s += src[h];
        val = s * SCALE;
    } else if (n < 256) {
        val = Wk[(size_t)k * QKV_DIM + (n - 128)];
    } else {
        val = Wv[(size_t)k * QKV_DIM + (n - 256)];
    }
    Wb[tid] = f2bf(val);
}

// ---------------------------------------------------------------------------
// Kernel 2: bias[384] = scale*sum_h bq | bk | bv
// ---------------------------------------------------------------------------
__global__ void prep_bias(const float* __restrict__ bq,
                          const float* __restrict__ bk,
                          const float* __restrict__ bv,
                          float* __restrict__ bias) {
    const int n = threadIdx.x;
    if (n >= N_TOT) return;
    float val;
    if (n < 128) {
        float s = 0.f;
        #pragma unroll
        for (int h = 0; h < N_HEADS; ++h) s += bq[n * N_HEADS + h];
        val = s * SCALE;
    } else if (n < 256) {
        val = bk[n - 128];
    } else {
        val = bv[n - 256];
    }
    bias[n] = val;
}

// ---------------------------------------------------------------------------
// Kernel 3: fused GEMM (WMMA bf16) + rank-1 softmax + head-broadcast output.
// WG = 256 threads = 8 waves. Each wave owns ALL 32 rows x a distinct 48-col
// N slice (3 n-tiles): every B block is loaded exactly once per WG, each B
// fragment feeds 2 WMMAs, and B fragments are ping-pong double-buffered
// across k-steps so WMMAs never wait on the loads just issued.
// ---------------------------------------------------------------------------
__global__ __launch_bounds__(256) void mqa_main(const float* __restrict__ x,
                                                const unsigned short* __restrict__ Wb,
                                                const float* __restrict__ bias,
                                                float* __restrict__ out) {
    // LDS overlay: xs (K-loop staging) and qkv (epilogue) never live together.
    __shared__ __align__(16) unsigned char smem[32 * 388 * sizeof(float)]; // 49664B
    unsigned short (*xs)[136] = (unsigned short (*)[136])smem;             // 8704B
    float          (*qkv)[388] = (float (*)[388])smem;

    const int tid  = threadIdx.x;
    const int lane = tid & 31;
    const int wave = tid >> 5;         // 0..7 -> 48-column N slice
    const int half = lane >> 4;
    const int l16  = lane & 15;
    const int nt0  = wave * 3;         // first of 3 n-tiles for this wave
    const int rowBase = blockIdx.x * 32;

    v8f acc[2][3];
    #pragma unroll
    for (int mi = 0; mi < 2; ++mi)
        #pragma unroll
        for (int t = 0; t < 3; ++t)
            acc[mi][t] = (v8f){0.f,0.f,0.f,0.f,0.f,0.f,0.f,0.f};

    const char* wbBase = (const char*)Wb + lane * 32;

    for (int kb = 0; kb < EMB_DIM / 128; ++kb) {
        // --- stage x[32][kb*128 .. +128) as bf16 into LDS -------------------
        #pragma unroll
        for (int it = 0; it < 4; ++it) {
            const int s   = tid + it * 256;       // 0..1023 float4 slots
            const int row = s >> 5;
            const int c4  = s & 31;
            const float* src = x + (size_t)(rowBase + row) * EMB_DIM + kb * 128 + c4 * 4;
            if (kb + 1 < EMB_DIM / 128)
                __builtin_prefetch(src + 128, 0, 1);   // global_prefetch next chunk
            const float4 f = *(const float4*)src;
            uint2 pk;
            pk.x = (unsigned)f2bf(f.x) | ((unsigned)f2bf(f.y) << 16);
            pk.y = (unsigned)f2bf(f.z) | ((unsigned)f2bf(f.w) << 16);
            *(uint2*)&xs[row][c4 * 4] = pk;
        }
        __syncthreads();

        // --- 4 WMMA k-steps of 32, B ping-pong double-buffered --------------
        Frag16 bb[2][3];
        #pragma unroll
        for (int t = 0; t < 3; ++t) {            // prologue: B for ki=0
            const uint4* bp = (const uint4*)(wbBase + (size_t)((nt0 + t) * 128 + kb * 4) * 1024);
            bb[0][t].q[0] = bp[0];
            bb[0][t].q[1] = bp[1];
        }
        #pragma unroll
        for (int ki = 0; ki < 4; ++ki) {
            if (ki < 3) {                         // issue next k-step's B loads
                #pragma unroll
                for (int t = 0; t < 3; ++t) {
                    const uint4* bp = (const uint4*)(wbBase +
                        (size_t)((nt0 + t) * 128 + kb * 4 + ki + 1) * 1024);
                    bb[(ki + 1) & 1][t].q[0] = bp[0];
                    bb[(ki + 1) & 1][t].q[1] = bp[1];
                }
            }
            const int k0 = ki * 32;
            Frag16 a[2];
            #pragma unroll
            for (int mi = 0; mi < 2; ++mi) {
                const unsigned short* arow = &xs[mi * 16 + l16][0];
                a[mi].q[0] = *(const uint4*)(arow + k0 + 8 * half);
                a[mi].q[1] = *(const uint4*)(arow + k0 + 16 + 8 * half);
            }
            #pragma unroll
            for (int mi = 0; mi < 2; ++mi)
                #pragma unroll
                for (int t = 0; t < 3; ++t)
                    acc[mi][t] = __builtin_amdgcn_wmma_f32_16x16x32_bf16(
                        false, a[mi].v, false, bb[ki & 1][t].v, (short)0,
                        acc[mi][t], false, false);
        }
        __syncthreads();     // xs reads done before restage (and before qkv overlay)
    }

    // --- D tiles + bias -> LDS qkv (C/D layout: row = 8*half + r, col = l16) ---
    #pragma unroll
    for (int mi = 0; mi < 2; ++mi) {
        #pragma unroll
        for (int t = 0; t < 3; ++t) {
            const int col  = wave * 48 + t * 16 + l16;
            const float bc = bias[col];
            #pragma unroll
            for (int r = 0; r < 8; ++r) {
                qkv[mi * 16 + 8 * half + r][col] = acc[mi][t][r] + bc;
            }
        }
    }
    __syncthreads();

    // --- rank-1 softmax epilogue: 8 threads per token, 16 d-rows each -------
    const int tok = tid >> 3;
    const int sub = tid & 7;
    const float* qrow = &qkv[tok][0];
    const float* kv   = &qkv[tok][128];
    const float* vv   = &qkv[tok][256];

    float kmax = -1e30f, kmin = 1e30f;
    for (int e = 0; e < 128; ++e) {
        const float ke = kv[e];
        kmax = fmaxf(kmax, ke);
        kmin = fminf(kmin, ke);
    }

    float* obase = out + (size_t)(rowBase + tok) * (QKV_DIM * N_HEADS);
    for (int dd = sub * 16; dd < sub * 16 + 16; ++dd) {
        const float a = qrow[dd];                    // scale already folded in
        const float m = (a >= 0.f) ? a * kmax : a * kmin;
        float s = 0.f, o = 0.f;
        for (int e = 0; e < 128; ++e) {
            const float w = __expf(a * kv[e] - m);
            s += w;
            o += w * vv[e];
        }
        const float val = o / s;
        const v4f f4 = {val, val, val, val};
        v4f* dst = (v4f*)(obase + dd * N_HEADS);         // broadcast over 32 heads
        #pragma unroll
        for (int j = 0; j < 8; ++j)
            __builtin_nontemporal_store(f4, dst + j);    // write-once: keep L2 for Wb/x
    }
}

// ---------------------------------------------------------------------------
extern "C" void kernel_launch(void* const* d_in, const int* in_sizes, int n_in,
                              void* d_out, int out_size, void* d_ws, size_t ws_size,
                              hipStream_t stream) {
    const float* x  = (const float*)d_in[0];
    const float* Wq = (const float*)d_in[1];
    const float* bq = (const float*)d_in[2];
    const float* Wk = (const float*)d_in[3];
    const float* bk = (const float*)d_in[4];
    const float* Wv = (const float*)d_in[5];
    const float* bv = (const float*)d_in[6];
    float* out = (float*)d_out;

    unsigned short* Wb = (unsigned short*)d_ws;                   // 4096*384*2 = 3MB
    float* bias = (float*)((char*)d_ws + (size_t)EMB_DIM * N_TOT * 2);

    const int M = in_sizes[0] / EMB_DIM;                          // B*S = 16384

    prep_wb  <<<(EMB_DIM * N_TOT) / 256, 256, 0, stream>>>(Wq, Wk, Wv, Wb);
    prep_bias<<<1, 512, 0, stream>>>(bq, bk, bv, bias);
    mqa_main <<<M / 32, 256, 0, stream>>>(x, Wb, bias, out);
}
